// _WaveRNNInference_29059748725557
// MI455X (gfx1250) — matrix operand order repaired and checked by reading, not compiled, not run on hardware
//
#include <hip/hip_runtime.h>
#include <hip/hip_bf16.h>

// ---------------------------------------------------------------------------
// WaveRNN inference for MI455X (gfx1250), persistent multi-WGP design, v2.
//  * 43 WGs x 512 threads. WGs 1..42: hidden matmul, 8 wave32s each = 336
//    waves <-> 168 m-tiles x 2 n-tiles, K-loop of 28 x v_wmma_f32_16x16x32_f16;
//    waves 8..15 prefetch cond[t+1] into L2.
//  * WG0 overlaps the matmul with gate-input precompute (cpre = input proj +
//    biases + conditional), which only depends on previous-step scalars. The
//    post-barrier serial tail is just gate fusion + bins WMMA + argmax.
//  * hp stored [b][m] so Phase-A accumulators store as contiguous v8f and the
//    tail reads coalesced.
//  * Weights pre-packed once into a per-lane A-fragment layout (contiguous k); hidden state
//    kept f32 (master) + f16 row-major (direct B-fragment 32B loads).
// ---------------------------------------------------------------------------

typedef __attribute__((ext_vector_type(16))) _Float16 v16h;
typedef __attribute__((ext_vector_type(8)))  float    v8f;

#define SIZE_  896
#define HALF_  448
#define BINS_  256
#define B_     32
#define T_     256
#define M_     2688          // 3*SIZE
#define MT_    168           // M_/16
#define KT_    28            // SIZE_/32
#define NWG_   43            // 1 tail/precompute WG + 42 matmul WGs
#define TPB_   512
#define GATE_  (B_ * HALF_)  // 14336 elements per gate plane

// workspace byte offsets (all 256B aligned)
#define WS_WPH   256                      // packed hidden weights f16: 168*28*512 h = 4,816,896 B
#define WS_WBC   (WS_WPH + 4816896)       // packed coarse-bins weights f16: 229,376 B
#define WS_WBF   (WS_WBC + 229376)        // packed fine-bins weights  f16: 229,376 B
#define WS_HP    (WS_WBF + 229376)        // hp[b][m] f32: 32*2688*4 = 344,064 B
#define WS_H16   (WS_HP + 344064)         // h16[b][896] f16 = 57,344 B
#define WS_HM    (WS_H16 + 57344)         // master hidden f32 [b][896] = 114,688 B
#define WS_PREV  (WS_HM + 114688)         // prev coarse[32], prev fine[32] f32 (256 B)
#define WS_CPRE  (WS_PREV + 256)          // cpre[6][b][448] f32 = 344,064 B

__device__ __forceinline__ float sigm(float x)      { return 1.0f / (1.0f + __expf(-x)); }
__device__ __forceinline__ float tanh_fast(float x) { return 1.0f - 2.0f / (1.0f + __expf(2.0f * x)); }

// ---------------------------------------------------------------------------
// Prep: zero barrier, init hidden state, pack weights to WMMA A-fragment order.
// A order used here (16-bit, 16x32, contiguous k): lane L -> row m = L%16; element e (0..15) of the
// lane's 32B chunk -> k = (L/16)*16 + e.
// ---------------------------------------------------------------------------
__global__ void __launch_bounds__(256) wavernn_prep(
    const float* __restrict__ W, const float* __restrict__ wbc_f32,
    const float* __restrict__ wbf_f32, const float* __restrict__ hstate,
    char* __restrict__ ws)
{
  size_t i = (size_t)blockIdx.x * blockDim.x + threadIdx.x;
  _Float16* wph = (_Float16*)(ws + WS_WPH);
  _Float16* wbc = (_Float16*)(ws + WS_WBC);
  _Float16* wbf = (_Float16*)(ws + WS_WBF);
  _Float16* h16 = (_Float16*)(ws + WS_H16);
  float*    hm  = (float*)(ws + WS_HM);
  float*    prv = (float*)(ws + WS_PREV);

  if (i < (size_t)M_ * SIZE_) {
    int e = (int)(i & 15), lane = (int)((i >> 4) & 31);
    int tile = (int)(i >> 9);
    int mt = tile / KT_, kt = tile - mt * KT_;
    int m = mt * 16 + (lane & 15);
    int k = kt * 32 + (lane >> 4) * 16 + e;
    wph[i] = (_Float16)W[(size_t)m * SIZE_ + k];
  }
  if (i < (size_t)16 * 14 * 512) {
    int e = (int)(i & 15), lane = (int)((i >> 4) & 31);
    int tile = (int)(i >> 9);
    int mt = tile / 14, kt = tile - mt * 14;
    int m = mt * 16 + (lane & 15);
    int k = kt * 32 + (lane >> 4) * 16 + e;
    wbc[i] = (_Float16)wbc_f32[m * HALF_ + k];
    wbf[i] = (_Float16)wbf_f32[m * HALF_ + k];
  }
  if (i < (size_t)B_ * SIZE_) { h16[i] = (_Float16)hstate[i]; hm[i] = hstate[i]; }
  if (i < B_) { prv[i] = 128.0f / 127.5f - 1.0f; prv[B_ + i] = -1.0f; }
  if (i == 0) { ((int*)ws)[0] = 0; ((int*)ws)[1] = 0; }
}

// ---------------------------------------------------------------------------
// Sense-reversal grid barrier across NWG_ resident workgroups.
// ---------------------------------------------------------------------------
__device__ __forceinline__ void grid_barrier(int* cnt, int* gen, int& lg)
{
  __syncthreads();
  if (threadIdx.x == 0) {
    int target = ++lg;
    if (__hip_atomic_fetch_add(cnt, 1, __ATOMIC_ACQ_REL, __HIP_MEMORY_SCOPE_AGENT) == NWG_ - 1) {
      __hip_atomic_store(cnt, 0, __ATOMIC_RELAXED, __HIP_MEMORY_SCOPE_AGENT);
      __hip_atomic_fetch_add(gen, 1, __ATOMIC_RELEASE, __HIP_MEMORY_SCOPE_AGENT);
    } else {
      while (__hip_atomic_load(gen, __ATOMIC_ACQUIRE, __HIP_MEMORY_SCOPE_AGENT) < target)
        __builtin_amdgcn_s_sleep(1);
    }
  }
  __syncthreads();
}

// ---------------------------------------------------------------------------
// Bins logits: (32 x 448) @ (448 x 256)^T with WMMA. WG0's 16 waves, one
// 16-bin m-tile per wave, both batch n-tiles, 14 K-tiles.
// ---------------------------------------------------------------------------
__device__ __forceinline__ void bins_matmul(const _Float16* __restrict__ wp,
                                            const _Float16* __restrict__ h16,
                                            int colbase,
                                            const float* __restrict__ bias,
                                            float* __restrict__ s_logits,
                                            int wave, int lane)
{
  const int mt = wave;
  const int brow0 = lane & 15;
  const int koff  = (lane >> 4) * 16;
  const _Float16* wprow = wp + (size_t)mt * 14 * 512 + lane * 16;
#pragma unroll
  for (int ntile = 0; ntile < 2; ++ntile) {
    v8f acc = {0.f, 0.f, 0.f, 0.f, 0.f, 0.f, 0.f, 0.f};
    const _Float16* hrow = h16 + (ntile * 16 + brow0) * SIZE_ + colbase;
    for (int kt = 0; kt < 14; ++kt) {
      v16h a = *(const v16h*)(wprow + kt * 512);
      v16h b = *(const v16h*)(hrow + kt * 32 + koff);
      acc = __builtin_amdgcn_wmma_f32_16x16x32_f16(false, a, false, b,
                                                   (short)0, acc, false, false);
    }
    int n  = ntile * 16 + brow0;
    int mb = mt * 16 + 8 * (lane >> 4);
#pragma unroll
    for (int r = 0; r < 8; ++r)
      s_logits[n * BINS_ + mb + r] = acc[r] + bias[mb + r];
  }
}

__device__ __forceinline__ void argmax_tail(const float* __restrict__ s_logits,
                                            float* __restrict__ s_red,
                                            int* __restrict__ s_redi,
                                            float* __restrict__ s_vals, int voff,
                                            int* __restrict__ outp, int t, int tid)
{
  int b = tid >> 4, seg = tid & 15;
  float best = -3.4e38f; int bi = 0;
#pragma unroll
  for (int k = 0; k < 16; ++k) {
    int bin = seg * 16 + k;
    float v = s_logits[b * BINS_ + bin];
    if (v > best) { best = v; bi = bin; }
  }
  s_red[tid] = best; s_redi[tid] = bi;
  __syncthreads();
  if (tid < B_) {
    float bb = -3.4e38f; int ii = 0;
    for (int s = 0; s < 16; ++s) {
      float v = s_red[tid * 16 + s];
      if (v > bb) { bb = v; ii = s_redi[tid * 16 + s]; }
    }
    outp[tid * T_ + t] = ii;                                  // argmax index
    s_vals[voff + tid] = (float)ii * (1.0f / 127.5f) - 1.0f;  // _scale(idx)
  }
  __syncthreads();
}

// ---------------------------------------------------------------------------
// Persistent main kernel: 43 WGs x 512 threads, loops the 256 timesteps.
// ---------------------------------------------------------------------------
__global__ void __launch_bounds__(TPB_) wavernn_main(
    const float* __restrict__ cond, const float* __restrict__ gib,
    const float* __restrict__ ghb,  const float* __restrict__ wci,
    const float* __restrict__ bci,  const float* __restrict__ wfi,
    const float* __restrict__ bfi,  const float* __restrict__ bbc,
    const float* __restrict__ bbf,  char* __restrict__ ws,
    int* __restrict__ outi)
{
  int* bar_cnt = (int*)ws;
  int* bar_gen = (int*)(ws + 4);
  const _Float16* wph = (const _Float16*)(ws + WS_WPH);
  const _Float16* wbc = (const _Float16*)(ws + WS_WBC);
  const _Float16* wbf = (const _Float16*)(ws + WS_WBF);
  float*    hp    = (float*)(ws + WS_HP);     // [b][2688]
  _Float16* h16   = (_Float16*)(ws + WS_H16); // [b][896]
  float*    hm    = (float*)(ws + WS_HM);     // [b][896]
  float*    prevc = (float*)(ws + WS_PREV);
  float*    prevf = prevc + B_;
  float*    cpre  = (float*)(ws + WS_CPRE);   // [6][b][448]
  int*   outc = outi;
  int*   outf = outi + B_ * T_;
  float* outh = (float*)(outi + 2 * B_ * T_);

  const int tid  = threadIdx.x;
  const int wave = tid >> 5, lane = tid & 31;
  const int wg   = blockIdx.x;
  int lg = 0;

  __shared__ float s_logits[B_ * BINS_];   // 32 KB
  __shared__ float s_red[TPB_];
  __shared__ int   s_redi[TPB_];
  __shared__ float s_vals[64];             // new coarse[32], new fine[32]

  // static tile assignment: WGs 1..42, waves 0..7 -> 336 waves <-> 168 mt x 2 nt
  const int p      = (wg - 1) * 8 + wave;
  const int mt     = p >> 1, nt = p & 1;
  const int brow   = nt * 16 + (lane & 15);
  const int koff   = (lane >> 4) * 16;
  const int ncol   = nt * 16 + (lane & 15);
  const int mbase0 = mt * 16 + 8 * (lane >> 4);

  for (int t = 0; t < T_; ++t) {
    // ---- Phase A -----------------------------------------------------------
    if (wg > 0) {
      if (wave < 8) {
        // hp[b][m] = (h @ W^T)[b][m] + gru_hidden_bias[m]
        v8f acc = {0.f, 0.f, 0.f, 0.f, 0.f, 0.f, 0.f, 0.f};
        const _Float16* arow = wph + (size_t)mt * KT_ * 512 + lane * 16;
        const _Float16* hrow = h16 + brow * SIZE_;
#pragma unroll 4
        for (int kt = 0; kt < KT_; ++kt) {
          v16h a = *(const v16h*)(arow + kt * 512);
          v16h b = *(const v16h*)(hrow + kt * 32 + koff);
          acc = __builtin_amdgcn_wmma_f32_16x16x32_f16(false, a, false, b,
                                                       (short)0, acc, false, false);
        }
        acc += *(const v8f*)(ghb + mbase0);                 // vector bias add
        *(v8f*)(hp + (size_t)ncol * M_ + mbase0) = acc;     // 2x b128, coalesced tail reads
      } else if (t + 1 < T_) {
        // waves 8..15 prefetch next step's conditional slab into L2
        int line = (wg - 1) * 256 + (tid - 256);    // 32*2688*4 B = 5376 lines
        if (line < 5376) {
          int b = line / 168, l = line - b * 168;
          __builtin_prefetch((const char*)cond +
                             ((size_t)(b * T_ + t + 1) * M_) * 4 + (size_t)l * 64, 0, 1);
        }
      }
    } else {
      // WG0: precompute gate inputs for step t (depends only on prev scalars).
      // cpre[gg][b][j], gg: 0=cr 1=cu 2=ce (coarse), 3=fr 4=fu 5=fe (fine,
      // missing the coarse_new*w term which is added in the tail).
      for (int i = tid; i < 6 * GATE_; i += TPB_) {
        int gg = i / GATE_;
        int r2 = i - gg * GATE_;
        int b = r2 / HALF_, j = r2 - b * HALF_;
        int fine = gg >= 3;
        int g = fine ? gg - 3 : gg;          // conditional group / gate row block
        int col = fine ? HALF_ + j : j;      // column within SIZE_
        int wrow = g * HALF_ + j;            // row of w_*_in / b_*_in
        float co = prevc[b], fi = prevf[b];
        float x = cond[(size_t)(b * T_ + t) * M_ + g * SIZE_ + col] + gib[g * SIZE_ + col];
        float v;
        if (!fine) v = co * wci[2 * wrow] + fi * wci[2 * wrow + 1] + bci[wrow] + x;
        else       v = co * wfi[3 * wrow] + fi * wfi[3 * wrow + 1] + bfi[wrow] + x;
        cpre[i] = v;
      }
    }
    __threadfence();
    grid_barrier(bar_cnt, bar_gen, lg);

    // ---- Phase B: serial tail on WG0 ---------------------------------------
    if (wg == 0) {
      // coarse gates: hcr=hp[b][j], hcu=hp[b][896+j], hce=hp[b][1792+j]
      for (int i = tid; i < GATE_; i += TPB_) {
        int b = i / HALF_, j = i - b * HALF_;
        const float* hpb = hp + (size_t)b * M_;
        float r = sigm(cpre[0 * GATE_ + i] + hpb[j]);
        float u = sigm(cpre[1 * GATE_ + i] + hpb[2 * HALF_ + j]);
        float e = tanh_fast(r * hpb[4 * HALF_ + j] + cpre[2 * GATE_ + i]);
        float chv = u * hm[b * SIZE_ + j] + (1.0f - u) * e;
        hm[b * SIZE_ + j] = chv;
        h16[b * SIZE_ + j] = (_Float16)chv;
      }
      __threadfence(); __syncthreads();
      bins_matmul(wbc, h16, 0, bbc, s_logits, wave, lane);
      __syncthreads();
      argmax_tail(s_logits, s_red, s_redi, s_vals, 0, outc, t, tid);

      // fine gates: hfr=hp[b][448+j], hfu=hp[b][1344+j], hfe=hp[b][2240+j]
      for (int i = tid; i < GATE_; i += TPB_) {
        int b = i / HALF_, j = i - b * HALF_;
        const float* hpb = hp + (size_t)b * M_;
        float cn = s_vals[b];
        float r = sigm(cpre[3 * GATE_ + i] + cn * wfi[3 * j + 2]              + hpb[HALF_ + j]);
        float u = sigm(cpre[4 * GATE_ + i] + cn * wfi[3 * (HALF_ + j) + 2]    + hpb[3 * HALF_ + j]);
        float e = tanh_fast(r * hpb[5 * HALF_ + j] +
                            cpre[5 * GATE_ + i] + cn * wfi[3 * (2 * HALF_ + j) + 2]);
        float fhv = u * hm[b * SIZE_ + HALF_ + j] + (1.0f - u) * e;
        hm[b * SIZE_ + HALF_ + j] = fhv;
        h16[b * SIZE_ + HALF_ + j] = (_Float16)fhv;
      }
      __threadfence(); __syncthreads();
      bins_matmul(wbf, h16, HALF_, bbf, s_logits, wave, lane);
      __syncthreads();
      argmax_tail(s_logits, s_red, s_redi, s_vals, 32, outf, t, tid);

      if (tid < B_) { prevc[tid] = s_vals[tid]; prevf[tid] = s_vals[32 + tid]; }
      if (t == T_ - 1) {
        for (int i = tid; i < B_ * SIZE_; i += TPB_) outh[i] = hm[i];
      }
      __threadfence();
    }
    grid_barrier(bar_cnt, bar_gen, lg);
  }
}

// ---------------------------------------------------------------------------
extern "C" void kernel_launch(void* const* d_in, const int* in_sizes, int n_in,
                              void* d_out, int out_size, void* d_ws, size_t ws_size,
                              hipStream_t stream) {
  (void)in_sizes; (void)n_in; (void)out_size; (void)ws_size;
  const float* cond   = (const float*)d_in[0];
  const float* hstate = (const float*)d_in[1];
  const float* gib    = (const float*)d_in[2];
  const float* W      = (const float*)d_in[3];
  const float* ghb    = (const float*)d_in[4];
  const float* wci    = (const float*)d_in[5];
  const float* bci    = (const float*)d_in[6];
  const float* wfi    = (const float*)d_in[7];
  const float* bfi    = (const float*)d_in[8];
  const float* wbc    = (const float*)d_in[9];
  const float* bbc    = (const float*)d_in[10];
  const float* wbf    = (const float*)d_in[11];
  const float* bbf    = (const float*)d_in[12];
  char* ws = (char*)d_ws;

  const long npack = (long)M_ * SIZE_;             // 2,408,448 threads cover everything
  const int  pblocks = (int)((npack + 255) / 256);
  wavernn_prep<<<pblocks, 256, 0, stream>>>(W, wbc, wbf, hstate, ws);
  wavernn_main<<<NWG_, TPB_, 0, stream>>>(cond, gib, ghb, wci, bci, wfi, bfi,
                                          bbc, bbf, ws, (int*)d_out);
}
